// VisionTransformer_21766894256180
// MI455X (gfx1250) — matrix operand order repaired and checked by reading, not compiled, not run on hardware
//
#include <hip/hip_runtime.h>
#include <stdint.h>

// ---------------- model dims ----------------
#define L_   12
#define H_   12
#define D_   768
#define HD_  64
#define MM_  3072
#define B_   32
#define NC_  1000
#define S_   577          // 576 patches + cls
#define SP_  608          // padded key stride for V^T rows (multiple of 32)
#define NP_  576
#define MROWS_ (B_*S_)    // 18464
#define SCALE_ 0.125f     // 1/sqrt(64)

#define GF_BIAS  1
#define GF_GELU  2
#define GF_RESID 4

typedef __attribute__((ext_vector_type(16))) __bf16 v16bf;
typedef __attribute__((ext_vector_type(8)))  float  v8f;

// ---------------- helpers ----------------
__device__ __forceinline__ uint16_t f2bf(float f){
  union { float f; uint32_t u; } c; c.f = f;
  return (uint16_t)((c.u + 0x7FFFu + ((c.u >> 16) & 1u)) >> 16);  // RNE
}

__device__ __forceinline__ v8f zero8(){
  v8f z;
#pragma unroll
  for (int i = 0; i < 8; ++i) z[i] = 0.f;
  return z;
}

__device__ __forceinline__ v8f wmma_bf16(v16bf a, v16bf b, v8f c){
  // (neg_a, A, neg_b, B, c_mod, C, reuse_a, reuse_b)
  return __builtin_amdgcn_wmma_f32_16x16x32_bf16(false, a, false, b, (short)0, c, false, false);
}

// A-matrix 16x32 bf16 fragment (ISA 7.12.2): lane&15 = M row;
// lane<16 -> K = {k0..k0+7, k0+16..k0+23}; lane>=16 -> K = {k0+8..k0+15, k0+24..k0+31}
__device__ __forceinline__ v16bf ldfragA(const uint16_t* base, int stride,
                                         int row0, int maxRow, int k0){
  const int lane = threadIdx.x & 31;
  int row = row0 + (lane & 15); if (row > maxRow) row = maxRow;
  const int hi = lane >> 4;
  const uint16_t* p = base + (size_t)row * (size_t)stride + k0 + hi*8;
  union { uint4 q[2]; v16bf v; } f;
  f.q[0] = *(const uint4*)p;          // 8 bf16: K run 0
  f.q[1] = *(const uint4*)(p + 16);   // 8 bf16: K run 1 (+16 elements)
  return f.v;
}

// B-matrix 32x16 bf16 fragment: lane&15 = N col (row of K-major Bt);
// lane<16 -> K = {k0..k0+7, k0+16..k0+23}; lane>=16 -> K = {k0+8..k0+15, k0+24..k0+31}
__device__ __forceinline__ v16bf ldfragB(const uint16_t* base, int stride,
                                         int row0, int maxRow, int k0){
  const int lane = threadIdx.x & 31;
  int row = row0 + (lane & 15); if (row > maxRow) row = maxRow;
  const int hi = lane >> 4;
  const uint16_t* p = base + (size_t)row * (size_t)stride + k0 + hi*8;
  union { uint4 q[2]; v16bf v; } f;
  f.q[0] = *(const uint4*)p;
  f.q[1] = *(const uint4*)(p + 16);
  return f.v;
}

// ---------------- elementwise kernels ----------------
__global__ void k_cvt_bf16(const float* __restrict__ in, uint16_t* __restrict__ out, long long n){
  long long i = (long long)blockIdx.x * blockDim.x + threadIdx.x;
  if (i < n) out[i] = f2bf(in[i]);
}

// in: [batch][K][N] fp32 -> out: [batch][N][K] bf16 (K-contiguous weights for WMMA B operand)
__global__ void k_transpose_cvt(const float* __restrict__ in, uint16_t* __restrict__ out,
                                int K, int N, long long n){
  long long i = (long long)blockIdx.x * blockDim.x + threadIdx.x;
  if (i >= n) return;
  long long per = (long long)K * N;
  long long b = i / per;
  long long r = i - b * per;
  int nn = (int)(r / K);
  int kk = (int)(r - (long long)nn * K);
  out[i] = f2bf(in[b * per + (long long)kk * N + nn]);
}

// A[b*576+p][k], p=py*24+px, k=c*256+ph*16+pw  (stride-16 conv == GEMM)
__global__ void k_im2col(const float* __restrict__ x, uint16_t* __restrict__ A){
  long long i = (long long)blockIdx.x * blockDim.x + threadIdx.x;
  long long n = (long long)B_ * NP_ * 768;
  if (i >= n) return;
  int kk = (int)(i % 768);
  int p  = (int)((i / 768) % NP_);
  int b  = (int)(i / (768LL * NP_));
  int py = p / 24, px = p % 24;
  int c  = kk / 256, r2 = kk % 256, ph = r2 / 16, pw = r2 % 16;
  A[i] = f2bf(x[(((size_t)b * 3 + c) * 384 + (py*16 + ph)) * 384 + (px*16 + pw)]);
}

// faithful torch .view(b,-1,D) channel-major remap + cls + pos_emb
__global__ void k_assemble(const float* __restrict__ convf, const float* __restrict__ cls,
                           const float* __restrict__ pos, float* __restrict__ x0){
  long long i = (long long)blockIdx.x * blockDim.x + threadIdx.x;
  long long n = (long long)B_ * S_ * 768;
  if (i >= n) return;
  int f = (int)(i % 768);
  int t = (int)((i / 768) % S_);
  int b = (int)(i / (768LL * S_));
  float v;
  if (t == 0) v = cls[f];
  else {
    int flat = (t - 1) * 768 + f;
    int d = flat / NP_;        // channel
    int pp = flat % NP_;       // spatial
    v = convf[((size_t)b * NP_ + pp) * 768 + d];
  }
  v += pos[(size_t)t * 768 + f];
  x0[i] = v;
}

// qkvf: [B*S, 2304] fp32 (n = h*192 + e; 0..63 q, 64..127 k, 128..191 v)
__global__ void k_split_qk(const float* __restrict__ C, uint16_t* __restrict__ q,
                           uint16_t* __restrict__ k){
  long long i = (long long)blockIdx.x * blockDim.x + threadIdx.x;
  long long n = (long long)B_ * H_ * S_ * HD_;
  if (i >= n) return;
  int e = (int)(i % HD_);
  int s = (int)((i / HD_) % S_);
  int h = (int)((i / ((long long)HD_ * S_)) % H_);
  int b = (int)(i / ((long long)HD_ * S_ * H_));
  size_t base = (size_t)(b * S_ + s) * 2304 + h * 192;
  q[i] = f2bf(C[base + e]);
  k[i] = f2bf(C[base + 64 + e]);
}

// V transposed: vt[B][H][e=0..63][key padded to SP_]; pad zeroed (keeps P*V NaN-free)
__global__ void k_split_v(const float* __restrict__ C, uint16_t* __restrict__ vt){
  long long i = (long long)blockIdx.x * blockDim.x + threadIdx.x;
  long long n = (long long)B_ * H_ * HD_ * SP_;
  if (i >= n) return;
  int sp = (int)(i % SP_);
  int e  = (int)((i / SP_) % HD_);
  int h  = (int)((i / ((long long)SP_ * HD_)) % H_);
  int b  = (int)(i / ((long long)SP_ * HD_ * H_));
  uint16_t v = 0;
  if (sp < S_) v = f2bf(C[(size_t)(b * S_ + sp) * 2304 + h * 192 + 128 + e]);
  vt[i] = v;
}

// ---------------- LayerNorm: one wave per 768-wide row ----------------
__global__ void k_layernorm(const float* __restrict__ X, const float* __restrict__ g,
                            const float* __restrict__ bb, float* __restrict__ Yf,
                            uint16_t* __restrict__ Yb, int rows,
                            long long strideIn, long long strideOutF, long long strideOutB){
  int wid = blockIdx.x * (blockDim.x >> 5) + (threadIdx.x >> 5);
  if (wid >= rows) return;
  int lane = threadIdx.x & 31;
  const float* x = X + (long long)wid * strideIn;
  float v[24]; float s = 0.f, s2 = 0.f;
#pragma unroll
  for (int i = 0; i < 24; ++i){ float t = x[lane + 32*i]; v[i] = t; s += t; s2 += t*t; }
#pragma unroll
  for (int m = 16; m; m >>= 1){ s += __shfl_xor(s, m, 32); s2 += __shfl_xor(s2, m, 32); }
  float mu  = s * (1.f/768.f);
  float var = s2 * (1.f/768.f) - mu*mu;
  float inv = rsqrtf(var + 1e-5f);
#pragma unroll
  for (int i = 0; i < 24; ++i){
    int c = lane + 32*i;
    float y = (v[i] - mu) * inv * g[c] + bb[c];
    if (Yf) Yf[(long long)wid*strideOutF + c] = y;
    if (Yb) Yb[(long long)wid*strideOutB + c] = f2bf(y);
  }
}

// ---------------- generic bf16 WMMA GEMM: C[M,N] = A[M,K] * Bt[N,K]^T ----------------
__global__ void __launch_bounds__(256)
k_gemm_nt(const uint16_t* __restrict__ A, const uint16_t* __restrict__ Bt,
          const float* __restrict__ bias, const float* __restrict__ resid,
          float* __restrict__ Cf, uint16_t* __restrict__ Cb,
          int Mdim, int Ndim, int Kdim, int flags){
  const int lane = threadIdx.x & 31;
  const int w    = threadIdx.x >> 5;                 // 8 waves: 2 (M) x 4 (N)
  const int m0   = blockIdx.y * 64  + (w & 1) * 32;
  const int n0   = blockIdx.x * 128 + (w >> 1) * 32;

  v8f acc00 = zero8(), acc01 = zero8(), acc10 = zero8(), acc11 = zero8();
  for (int k0 = 0; k0 < Kdim; k0 += 32){
    v16bf a0 = ldfragA(A,  Kdim, m0,      Mdim-1, k0);
    v16bf a1 = ldfragA(A,  Kdim, m0 + 16, Mdim-1, k0);
    v16bf b0 = ldfragB(Bt, Kdim, n0,      Ndim-1, k0);
    v16bf b1 = ldfragB(Bt, Kdim, n0 + 16, Ndim-1, k0);
    acc00 = wmma_bf16(a0, b0, acc00);
    acc01 = wmma_bf16(a0, b1, acc01);
    acc10 = wmma_bf16(a1, b0, acc10);
    acc11 = wmma_bf16(a1, b1, acc11);
  }

  const int hi = lane >> 4, col = lane & 15;
  v8f tiles[4] = {acc00, acc01, acc10, acc11};
#pragma unroll
  for (int mi = 0; mi < 2; ++mi)
#pragma unroll
  for (int ni = 0; ni < 2; ++ni){
    v8f a = tiles[mi*2 + ni];
    int colb = n0 + ni*16 + col;
    int rowb = m0 + mi*16 + hi*8;
    float bv = (flags & GF_BIAS) ? bias[colb < Ndim ? colb : (Ndim-1)] : 0.f;
#pragma unroll
    for (int r = 0; r < 8; ++r){
      int row = rowb + r;
      if (row < Mdim && colb < Ndim){
        float vsum = a[r] + bv;
        if (flags & GF_GELU) vsum = 0.5f * vsum * (1.f + erff(vsum * 0.70710678118f));
        long long idx = (long long)row * Ndim + colb;
        if (flags & GF_RESID) vsum += resid[idx];
        if (Cf) Cf[idx] = vsum;
        if (Cb) Cb[idx] = f2bf(vsum);
      }
    }
  }
}

// ---------------- fused flash attention: one wave per 16-row q tile ----------------
__global__ void __launch_bounds__(32)
k_attention(const uint16_t* __restrict__ Qm, const uint16_t* __restrict__ Km,
            const uint16_t* __restrict__ Vt, uint16_t* __restrict__ Obf){
  const int bh = blockIdx.x;           // b*H + h
  const int b  = bh / H_, h = bh % H_;
  const int q0 = blockIdx.y * 16;
  const int lane = threadIdx.x & 31;
  const int col  = lane & 15;
  const int hi   = lane >> 4;

  const uint16_t* Q = Qm + (size_t)bh * S_ * HD_;
  const uint16_t* K = Km + (size_t)bh * S_ * HD_;
  const uint16_t* V = Vt + (size_t)bh * HD_ * SP_;

  // Q fragments resident (rows q0..q0+15, e in [0,32) and [32,64))
  v16bf qa0 = ldfragA(Q, HD_, q0, S_-1, 0);
  v16bf qa1 = ldfragA(Q, HD_, q0, S_-1, 32);

  float rmax[8], rsum[8];
#pragma unroll
  for (int r = 0; r < 8; ++r){ rmax[r] = -1e30f; rsum[r] = 0.f; }
  v8f o[4];
#pragma unroll
  for (int t = 0; t < 4; ++t) o[t] = zero8();

  __shared__ uint16_t Pl[16 * 40];     // P tile bounce (C-layout -> A-fragment), padded stride

  for (int kb = 0; kb < 19; ++kb){     // 19 * 32 = 608 keys (>= 577)
    const int kbase = kb * 32;
    // --- scores: S = Q * K^T for two 16-key tiles ---
    v8f s[2];
#pragma unroll
    for (int t = 0; t < 2; ++t){
      v16bf kf0 = ldfragB(K, HD_, kbase + t*16, S_-1, 0);
      v16bf kf1 = ldfragB(K, HD_, kbase + t*16, S_-1, 32);
      v8f acc = zero8();
      acc = wmma_bf16(qa0, kf0, acc);
      acc = wmma_bf16(qa1, kf1, acc);
      s[t] = acc;
    }
    // --- scale + key mask, tile row-max ---
    float tmax[8];
#pragma unroll
    for (int r = 0; r < 8; ++r) tmax[r] = -1e30f;
#pragma unroll
    for (int t = 0; t < 2; ++t){
      bool valid = (kbase + t*16 + col) < S_;
#pragma unroll
      for (int r = 0; r < 8; ++r){
        float v = valid ? s[t][r] * SCALE_ : -1e30f;
        s[t][r] = v;
        tmax[r] = fmaxf(tmax[r], v);
      }
    }
    // --- online softmax update (rows live per half-wave; xor<16 stays in half) ---
#pragma unroll
    for (int r = 0; r < 8; ++r){
      float m = tmax[r];
      m = fmaxf(m, __shfl_xor(m, 1, 32));
      m = fmaxf(m, __shfl_xor(m, 2, 32));
      m = fmaxf(m, __shfl_xor(m, 4, 32));
      m = fmaxf(m, __shfl_xor(m, 8, 32));
      float nm = fmaxf(rmax[r], m);
      float f  = __expf(rmax[r] - nm);
      rmax[r] = nm; rsum[r] *= f;
      o[0][r] *= f; o[1][r] *= f; o[2][r] *= f; o[3][r] *= f;
    }
    // --- P = exp(S - max); accumulate row sums; stash P tile to LDS ---
#pragma unroll
    for (int r = 0; r < 8; ++r){
      float p0 = __expf(s[0][r] - rmax[r]);
      float p1 = __expf(s[1][r] - rmax[r]);
      float ps = p0 + p1;
      ps += __shfl_xor(ps, 1, 32); ps += __shfl_xor(ps, 2, 32);
      ps += __shfl_xor(ps, 4, 32); ps += __shfl_xor(ps, 8, 32);
      rsum[r] += ps;
      int row = r + hi*8;
      Pl[row*40 + col]      = f2bf(p0);
      Pl[row*40 + 16 + col] = f2bf(p1);
    }
    asm volatile("s_wait_dscnt 0x0" ::: "memory");   // in-wave LDS write->read turnaround
    // --- reload P as 16x32 A-fragment ---
    union { uint4 q[2]; v16bf v; } pf;
    pf.q[0] = *(const uint4*)&Pl[col*40 + hi*8];
    pf.q[1] = *(const uint4*)&Pl[col*40 + hi*8 + 16];
    // --- O += P * V  (V pre-transposed: rows = e, K-contiguous keys) ---
#pragma unroll
    for (int t = 0; t < 4; ++t){
      v16bf vf = ldfragB(V, SP_, t*16, HD_-1, kbase);
      o[t] = wmma_bf16(pf.v, vf, o[t]);
    }
  }
  // --- normalize + store concat-head bf16 output ---
#pragma unroll
  for (int r = 0; r < 8; ++r){
    float inv = 1.f / rsum[r];
    int row = q0 + r + hi*8;
    if (row < S_){
#pragma unroll
      for (int t = 0; t < 4; ++t){
        int e = t*16 + col;
        Obf[(size_t)(b * S_ + row) * D_ + h * HD_ + e] = f2bf(o[t][r] * inv);
      }
    }
  }
}

// ---------------- head softmax: one wave per row ----------------
__global__ void k_softmax_rows(const float* __restrict__ in, float* __restrict__ out, int N){
  int r = blockIdx.x, lane = threadIdx.x;
  const float* x = in + (size_t)r * N;
  float mx = -3.0e38f;
  for (int c = lane; c < N; c += 32) mx = fmaxf(mx, x[c]);
  for (int m = 16; m; m >>= 1) mx = fmaxf(mx, __shfl_xor(mx, m, 32));
  float s = 0.f;
  for (int c = lane; c < N; c += 32) s += __expf(x[c] - mx);
  for (int m = 16; m; m >>= 1) s += __shfl_xor(s, m, 32);
  float inv = 1.f / s;
  for (int c = lane; c < N; c += 32) out[(size_t)r * N + c] = __expf(x[c] - mx) * inv;
}

// ---------------- host orchestration ----------------
extern "C" void kernel_launch(void* const* d_in, const int* in_sizes, int n_in,
                              void* d_out, int out_size, void* d_ws, size_t ws_size,
                              hipStream_t stream){
  (void)in_sizes; (void)n_in; (void)out_size; (void)ws_size;
  const float* x_img  = (const float*)d_in[0];
  const float* conv_w = (const float*)d_in[1];
  const float* conv_b = (const float*)d_in[2];
  const float* cls_tk = (const float*)d_in[3];
  const float* pos_e  = (const float*)d_in[4];
  const float* ln1_g  = (const float*)d_in[5];
  const float* ln1_b  = (const float*)d_in[6];
  const float* wqkv   = (const float*)d_in[7];
  const float* bqkv   = (const float*)d_in[8];
  const float* wmsa   = (const float*)d_in[9];
  const float* bmsa   = (const float*)d_in[10];
  const float* ln2_g  = (const float*)d_in[11];
  const float* ln2_b  = (const float*)d_in[12];
  const float* lnm_g  = (const float*)d_in[13];
  const float* lnm_b  = (const float*)d_in[14];
  const float* w1     = (const float*)d_in[15];
  const float* b1     = (const float*)d_in[16];
  const float* w2     = (const float*)d_in[17];
  const float* b2     = (const float*)d_in[18];
  const float* lnf_g  = (const float*)d_in[19];
  const float* lnf_b  = (const float*)d_in[20];
  const float* whead  = (const float*)d_in[21];
  const float* bhead  = (const float*)d_in[22];
  float* out = (float*)d_out;

  // ---- workspace carve (256B aligned) ----
  char* ws = (char*)d_ws;
  size_t cur = 0;
  auto carve = [&](size_t bytes)->char*{
    char* p = ws + cur; cur += (bytes + 255) & ~(size_t)255; return p;
  };
  uint16_t* wqkv_b  = (uint16_t*)carve((size_t)L_ * 2304 * 768 * 2);
  uint16_t* wmsa_b  = (uint16_t*)carve((size_t)L_ * 768 * 768 * 2);
  uint16_t* w1_b    = (uint16_t*)carve((size_t)L_ * 3072 * 768 * 2);
  uint16_t* w2_b    = (uint16_t*)carve((size_t)L_ * 768 * 3072 * 2);
  uint16_t* whead_b = (uint16_t*)carve((size_t)NC_ * 768 * 2);
  uint16_t* convw_b = (uint16_t*)carve((size_t)768 * 768 * 2);
  uint16_t* Apatch  = (uint16_t*)carve((size_t)B_ * NP_ * 768 * 2);
  float*    convf   = (float*)   carve((size_t)B_ * NP_ * 768 * 4);
  float*    xf      = (float*)   carve((size_t)MROWS_ * 768 * 4);
  float*    hf      = (float*)   carve((size_t)MROWS_ * 768 * 4);
  uint16_t* hb      = (uint16_t*)carve((size_t)MROWS_ * 768 * 2);
  float*    x1f     = (float*)   carve((size_t)MROWS_ * 768 * 4);
  float*    qkvf    = (float*)   carve((size_t)MROWS_ * 2304 * 4);
  uint16_t* qb      = (uint16_t*)carve((size_t)B_ * H_ * S_ * HD_ * 2);
  uint16_t* kbm     = (uint16_t*)carve((size_t)B_ * H_ * S_ * HD_ * 2);
  uint16_t* vtb     = (uint16_t*)carve((size_t)B_ * H_ * HD_ * SP_ * 2);
  uint16_t* obf     = (uint16_t*)carve((size_t)MROWS_ * 768 * 2);
  uint16_t* midb    = (uint16_t*)carve((size_t)MROWS_ * 3072 * 2);
  float*    clsf    = (float*)   carve((size_t)B_ * 768 * 4);
  uint16_t* clsb    = (uint16_t*)carve((size_t)B_ * 768 * 2);
  float*    headf   = (float*)   carve((size_t)B_ * NC_ * 4);

  auto g1 = [](long long n){ return dim3((unsigned)((n + 255) / 256)); };

  // ---- weight prep: fp32 -> bf16, transposed to [N][K] ----
  { long long n = (long long)L_*H_*768*192; k_transpose_cvt<<<g1(n),256,0,stream>>>(wqkv, wqkv_b, 768, 192, n); }
  { long long n = (long long)L_*768*768;    k_transpose_cvt<<<g1(n),256,0,stream>>>(wmsa, wmsa_b, 768, 768, n); }
  { long long n = (long long)L_*3072*768;   k_transpose_cvt<<<g1(n),256,0,stream>>>(w1,   w1_b,   768, 3072, n); }
  { long long n = (long long)L_*768*3072;   k_transpose_cvt<<<g1(n),256,0,stream>>>(w2,   w2_b,   3072, 768, n); }
  { long long n = (long long)768*NC_;       k_transpose_cvt<<<g1(n),256,0,stream>>>(whead, whead_b, 768, NC_, n); }
  { long long n = (long long)768*768;       k_cvt_bf16<<<g1(n),256,0,stream>>>(conv_w, convw_b, n); }

  // ---- patch embed (stride-16 conv as GEMM) + token assembly ----
  { long long n = (long long)B_*NP_*768; k_im2col<<<g1(n),256,0,stream>>>(x_img, Apatch); }
  { dim3 grid((768 + 127)/128, (B_*NP_ + 63)/64);
    k_gemm_nt<<<grid,256,0,stream>>>(Apatch, convw_b, conv_b, nullptr, convf, nullptr,
                                     B_*NP_, 768, 768, GF_BIAS); }
  { long long n = (long long)B_*S_*768; k_assemble<<<g1(n),256,0,stream>>>(convf, cls_tk, pos_e, xf); }

  // ---- encoder layers ----
  dim3 gLN((MROWS_ + 7) / 8);  // 8 waves / block, wave per row
  for (int l = 0; l < L_; ++l){
    // h = LN1(x) -> hf (fp32, residual source) + hb (bf16, GEMM operand)
    k_layernorm<<<gLN,256,0,stream>>>(xf, ln1_g + l*768, ln1_b + l*768, hf, hb,
                                      MROWS_, 768, 768, 768);
    // qkv = h @ Wqkv + b
    { dim3 grid((2304 + 127)/128, (MROWS_ + 63)/64);
      k_gemm_nt<<<grid,256,0,stream>>>(hb, wqkv_b + (size_t)l*2304*768, bqkv + l*2304,
                                       nullptr, qkvf, nullptr, MROWS_, 2304, 768, GF_BIAS); }
    { long long n = (long long)B_*H_*S_*HD_;  k_split_qk<<<g1(n),256,0,stream>>>(qkvf, qb, kbm); }
    { long long n = (long long)B_*H_*HD_*SP_; k_split_v<<<g1(n),256,0,stream>>>(qkvf, vtb); }
    // fused attention -> concat-head bf16
    { dim3 grid(B_*H_, (S_ + 15)/16); k_attention<<<grid,32,0,stream>>>(qb, kbm, vtb, obf); }
    // x1 = h + (o @ Wmsa + b)   (residual around MSA is post-LN1, faithful)
    { dim3 grid((768 + 127)/128, (MROWS_ + 63)/64);
      k_gemm_nt<<<grid,256,0,stream>>>(obf, wmsa_b + (size_t)l*768*768, bmsa + l*768,
                                       hf, x1f, nullptr, MROWS_, 768, 768, GF_BIAS|GF_RESID); }
    // h2 = LN_m(LN_2(x1))  (double LN, faithful); second LN reads hf, writes bf16 hb
    k_layernorm<<<gLN,256,0,stream>>>(x1f, ln2_g + l*768, ln2_b + l*768, hf, nullptr,
                                      MROWS_, 768, 768, 768);
    k_layernorm<<<gLN,256,0,stream>>>(hf, lnm_g + l*768, lnm_b + l*768, nullptr, hb,
                                      MROWS_, 768, 768, 768);
    // mid = gelu(h2 @ W1 + b1) -> bf16 directly
    { dim3 grid((3072 + 127)/128, (MROWS_ + 63)/64);
      k_gemm_nt<<<grid,256,0,stream>>>(hb, w1_b + (size_t)l*3072*768, b1 + l*3072,
                                       nullptr, nullptr, midb, MROWS_, 3072, 768, GF_BIAS|GF_GELU); }
    // x = x1 + (mid @ W2 + b2)
    { dim3 grid((768 + 127)/128, (MROWS_ + 63)/64);
      k_gemm_nt<<<grid,256,0,stream>>>(midb, w2_b + (size_t)l*768*3072, b2 + l*768,
                                       x1f, xf, nullptr, MROWS_, 768, 3072, GF_BIAS|GF_RESID); }
  }

  // ---- head: LN(cls) @ whead + bhead, softmax ----
  k_layernorm<<<dim3((B_ + 7)/8),256,0,stream>>>(xf, lnf_g, lnf_b, clsf, clsb,
                                                 B_, (long long)S_*768, 768, 768);
  { dim3 grid((NC_ + 127)/128, (B_ + 63)/64);
    k_gemm_nt<<<grid,256,0,stream>>>(clsb, whead_b, bhead, nullptr, headf, nullptr,
                                     B_, NC_, 768, GF_BIAS); }
  k_softmax_rows<<<dim3(B_),32,0,stream>>>(headf, out, NC_);
}
